// ApproximatedEMDLoss_19980187861538
// MI455X (gfx1250) — matrix operand order repaired and checked by reading, not compiled, not run on hardware
//
#include <hip/hip_runtime.h>
#include <hip/hip_bf16.h>

#define NN 2048
#define BB 8
#define EPS 0.1f

typedef __attribute__((ext_vector_type(2)))  float  v2f;
typedef __attribute__((ext_vector_type(8)))  float  v8f;
typedef __attribute__((ext_vector_type(16))) __bf16 v16bf;

union BF16F {
    v16bf v;
    uint4 q[2];
};

__device__ __forceinline__ unsigned short f2bf(float f) {
    unsigned int u = __float_as_uint(f);
    u += 0x7fffu + ((u >> 16) & 1u);          // round-to-nearest-even
    return (unsigned short)(u >> 16);
}
__device__ __forceinline__ float bf2f(unsigned short s) {
    return __uint_as_float(((unsigned int)s) << 16);
}
__device__ __forceinline__ float fast_sqrt(float x) {
    return __builtin_amdgcn_sqrtf(x);         // raw v_sqrt_f32, no fixup chain
}

// ---------------------------------------------------------------------------
// Kernel 1: build K (bf16) and K^T (bf16).  Gram matrix x.y^T via
// V_WMMA_F32_16X16X4_F32 (K padded 3->4 with zeros).
// Waves: B * (N/16 row tiles) * 4 col groups = 4096 waves, 512 blocks x 256.
// ---------------------------------------------------------------------------
__global__ void build_k_kernel(const float* __restrict__ x, const float* __restrict__ y,
                               unsigned short* __restrict__ Kb, unsigned short* __restrict__ KTb) {
    const int wv   = (blockIdx.x << 3) + (threadIdx.x >> 5);
    const int b    = wv >> 9;
    const int rt   = (wv >> 2) & 127;
    const int cg   = wv & 3;
    const int lane = threadIdx.x & 31;
    const int lm   = lane & 15;
    const int hi   = lane >> 4;

    const int n = rt * 16 + lm;
    const float* xr = x + ((size_t)b * NN + n) * 3;
    v2f a;
    a.x = xr[hi * 2];                 // K=0 (lo) / K=2 (hi)
    a.y = hi ? 0.f : xr[1];           // K=1 (lo) / K=3 pad (hi)
    const float xxn = xr[0]*xr[0] + xr[1]*xr[1] + xr[2]*xr[2];

    float xs[8];
#pragma unroll
    for (int r = 0; r < 8; ++r) xs[r] = __shfl(xxn, hi * 8 + r, 32);

    const int m_begin = cg * 512, m_end = m_begin + 512;
    for (int m0 = m_begin; m0 < m_end; m0 += 16) {
        const float* yr = y + ((size_t)b * NN + m0 + lm) * 3;
        v2f bb;
        bb.x = yr[hi * 2];
        bb.y = hi ? 0.f : yr[1];
        const float yy = yr[0]*yr[0] + yr[1]*yr[1] + yr[2]*yr[2];

        v8f c = {};
        c = __builtin_amdgcn_wmma_f32_16x16x4_f32(false, a, false, bb, (short)0, c, false, false);

        unsigned short kt[8];
#pragma unroll
        for (int r = 0; r < 8; ++r) {
            float d2 = xs[r] + yy - 2.0f * c[r];
            float dd = fast_sqrt(fmaxf(d2, 1e-12f));
            float kv = __expf(dd * (-1.0f / EPS));
            kt[r] = f2bf(kv);
            // K[n][m], n = rt*16 + hi*8 + r, m = m0+lm
            Kb[((size_t)b * NN + rt * 16 + hi * 8 + r) * NN + m0 + lm] = kt[r];
        }
        // K^T[m][n]: contiguous 8 elements per lane -> one 16B store
        uint4 pk;
        pk.x = (unsigned)kt[0] | ((unsigned)kt[1] << 16);
        pk.y = (unsigned)kt[2] | ((unsigned)kt[3] << 16);
        pk.z = (unsigned)kt[4] | ((unsigned)kt[5] << 16);
        pk.w = (unsigned)kt[6] | ((unsigned)kt[7] << 16);
        *(uint4*)(KTb + ((size_t)b * NN + m0 + lm) * NN + rt * 16 + hi * 8) = pk;
    }
}

// ---------------------------------------------------------------------------
// Kernel 2: colsum[b][m] = sum_n K[b][n][m].  Two columns per thread via b32
// loads -> full 128B line per wave load instruction.  Fixed order, no atomics.
// ---------------------------------------------------------------------------
__global__ void colsum_kernel(const unsigned short* __restrict__ Kb, float* __restrict__ cs) {
    const int idx = blockIdx.x * blockDim.x + threadIdx.x;   // 8192 threads
    const int b = idx >> 10, m2 = idx & 1023;                // column pair
    const unsigned int* p = (const unsigned int*)(Kb + (size_t)b * NN * NN) + m2;
    float s0 = 0.f, s1 = 0.f;
    for (int n = 0; n < NN; ++n) {
        unsigned int w = p[(size_t)n * (NN / 2)];
        s0 += __uint_as_float(w << 16);
        s1 += __uint_as_float(w & 0xffff0000u);
    }
    cs[b * NN + 2 * m2]     = s0;
    cs[b * NN + 2 * m2 + 1] = s1;
}

// w0 = v0/colsum with v0 = 1
__global__ void init_w_kernel(const float* __restrict__ cs, unsigned short* __restrict__ wbf) {
    const int idx = blockIdx.x * blockDim.x + threadIdx.x;
    wbf[idx] = f2bf(1.0f / cs[idx]);
}

// ---------------------------------------------------------------------------
// Sinkhorn GEMVs via V_WMMA_F32_16X16X32_BF16.  Wave pairs split the
// k-dimension (k in [0,1024) / [1024,2048)) and combine through LDS:
// 2048 waves -> 256 blocks x 256 threads, doubling load streams in flight.
// B operand is the vector replicated across the 16 columns.
// ---------------------------------------------------------------------------
#define GEMV_BODY(MAT, VEC)                                                     \
    const int wid  = threadIdx.x >> 5;                                          \
    const int pair = wid >> 1;                                                  \
    const int half = wid & 1;                                                   \
    const int gw   = blockIdx.x * 4 + pair;                                     \
    const int b    = gw >> 7;                                                   \
    const int rt   = gw & 127;                                                  \
    const int lane = threadIdx.x & 31;                                          \
    const int lm   = lane & 15;                                                 \
    const int hi   = lane >> 4;                                                 \
    const unsigned short* Ar = (MAT) + ((size_t)b * NN + rt * 16 + lm) * NN + hi * 8; \
    const unsigned short* wr = (VEC) + b * NN + hi * 8;                         \
    const int kbase = half * (NN / 2);                                          \
    v8f c = {};                                                                 \
    _Pragma("unroll 4")                                                         \
    for (int k0 = kbase; k0 < kbase + NN / 2; k0 += 32) {                       \
        BF16F af, bf;                                                           \
        af.q[0] = *(const uint4*)(Ar + k0);                                     \
        af.q[1] = *(const uint4*)(Ar + k0 + 16);                                \
        bf.q[0] = *(const uint4*)(wr + k0);                                     \
        bf.q[1] = *(const uint4*)(wr + k0 + 16);                                \
        c = __builtin_amdgcn_wmma_f32_16x16x32_bf16(false, af.v, false, bf.v,   \
                                                    (short)0, c, false, false); \
    }                                                                           \
    __shared__ float red[4 * 256];                                              \
    if (half) {                                                                 \
        _Pragma("unroll")                                                       \
        for (int r = 0; r < 8; ++r) red[pair * 256 + lane * 8 + r] = c[r];      \
    }                                                                           \
    __syncthreads();

__global__ void gemv_u_kernel(const unsigned short* __restrict__ Kb,
                              const unsigned short* __restrict__ wbf,
                              float* __restrict__ u, unsigned short* __restrict__ ubf) {
    GEMV_BODY(Kb, wbf)
    if (!half && lm == 0) {      // lanes 0 (rows 0-7) and 16 (rows 8-15)
#pragma unroll
        for (int r = 0; r < 8; ++r) {
            const float dot = c[r] + red[pair * 256 + lane * 8 + r];
            const float val = (1.0f / NN) / dot;
            const int row = b * NN + rt * 16 + hi * 8 + r;
            u[row]   = val;
            ubf[row] = f2bf(val);
        }
    }
}

__global__ void gemv_v_kernel(const unsigned short* __restrict__ KTb,
                              const unsigned short* __restrict__ ubf,
                              const float* __restrict__ cs,
                              float* __restrict__ v, unsigned short* __restrict__ wbf) {
    GEMV_BODY(KTb, ubf)
    if (!half && lm == 0) {
#pragma unroll
        for (int r = 0; r < 8; ++r) {
            const float dot = c[r] + red[pair * 256 + lane * 8 + r];
            const int m   = b * NN + rt * 16 + hi * 8 + r;
            const float inv = (1.0f / NN) / dot;    // = v[m]/colsum[m] = next w[m]
            v[m]   = cs[m] * inv;
            wbf[m] = f2bf(inv);
        }
    }
}

// ---------------------------------------------------------------------------
// Loss: recompute C,K tiles (f32, WMMA gram) and accumulate u[n]*K*v[m]*C.
// Same 4096-wave decomposition as build_k; per-wave partials, no atomics.
// ---------------------------------------------------------------------------
__global__ void loss_kernel(const float* __restrict__ x, const float* __restrict__ y,
                            const float* __restrict__ u, const float* __restrict__ v,
                            float* __restrict__ partials) {
    const int wv   = (blockIdx.x << 3) + (threadIdx.x >> 5);
    const int b    = wv >> 9;
    const int rt   = (wv >> 2) & 127;
    const int cg   = wv & 3;
    const int lane = threadIdx.x & 31;
    const int lm   = lane & 15;
    const int hi   = lane >> 4;

    const int n = rt * 16 + lm;
    const float* xr = x + ((size_t)b * NN + n) * 3;
    v2f a;
    a.x = xr[hi * 2];
    a.y = hi ? 0.f : xr[1];
    const float xxn  = xr[0]*xr[0] + xr[1]*xr[1] + xr[2]*xr[2];
    const float uval = u[b * NN + n];

    float xs[8], us[8];
#pragma unroll
    for (int r = 0; r < 8; ++r) {
        xs[r] = __shfl(xxn,  hi * 8 + r, 32);
        us[r] = __shfl(uval, hi * 8 + r, 32);
    }

    float sum = 0.f;
    const int m_begin = cg * 512, m_end = m_begin + 512;
    for (int m0 = m_begin; m0 < m_end; m0 += 16) {
        const float* yr = y + ((size_t)b * NN + m0 + lm) * 3;
        v2f bb;
        bb.x = yr[hi * 2];
        bb.y = hi ? 0.f : yr[1];
        const float yy = yr[0]*yr[0] + yr[1]*yr[1] + yr[2]*yr[2];
        const float vv = v[b * NN + m0 + lm];

        v8f c = {};
        c = __builtin_amdgcn_wmma_f32_16x16x4_f32(false, a, false, bb, (short)0, c, false, false);
#pragma unroll
        for (int r = 0; r < 8; ++r) {
            float d2 = xs[r] + yy - 2.0f * c[r];
            float dd = fast_sqrt(fmaxf(d2, 1e-12f));
            float kv = __expf(dd * (-1.0f / EPS));
            sum += us[r] * vv * kv * dd;
        }
    }
#pragma unroll
    for (int o = 16; o > 0; o >>= 1) sum += __shfl_xor(sum, o, 32);
    if (lane == 0) partials[wv] = sum;
}

__global__ void reduce_kernel(const float* __restrict__ partials, float* __restrict__ out) {
    __shared__ float sm[1024];
    const int t = threadIdx.x;
    float s = partials[t] + partials[t + 1024] + partials[t + 2048] + partials[t + 3072];
    sm[t] = s;
    __syncthreads();
    for (int o = 512; o > 0; o >>= 1) {
        if (t < o) sm[t] += sm[t + o];
        __syncthreads();
    }
    if (t == 0) out[0] = sm[0] / (float)BB;
}

// ---------------------------------------------------------------------------
extern "C" void kernel_launch(void* const* d_in, const int* in_sizes, int n_in,
                              void* d_out, int out_size, void* d_ws, size_t ws_size,
                              hipStream_t stream) {
    const float* x = (const float*)d_in[0];
    const float* y = (const float*)d_in[1];
    float* out = (float*)d_out;

    // workspace layout
    char* ws = (char*)d_ws;
    unsigned short* Kb  = (unsigned short*)ws;                                // 67,108,864 B
    unsigned short* KTb = (unsigned short*)(ws + (size_t)67108864);           // 67,108,864 B
    float* cs  = (float*)(ws + (size_t)134217728);                            // 64 KB
    float* u   = cs + BB * NN;
    float* v   = u  + BB * NN;
    unsigned short* ubf = (unsigned short*)(v + BB * NN);
    unsigned short* wbf = ubf + BB * NN;
    float* partials = (float*)(wbf + BB * NN);                                // 16 KB

    build_k_kernel<<<512, 256, 0, stream>>>(x, y, Kb, KTb);
    colsum_kernel<<<32, 256, 0, stream>>>(Kb, cs);
    init_w_kernel<<<64, 256, 0, stream>>>(cs, wbf);

    for (int it = 0; it < 50; ++it) {
        gemv_u_kernel<<<256, 256, 0, stream>>>(Kb, wbf, u, ubf);
        gemv_v_kernel<<<256, 256, 0, stream>>>(KTb, ubf, cs, v, wbf);
    }

    loss_kernel<<<512, 256, 0, stream>>>(x, y, u, v, partials);
    reduce_kernel<<<1, 1024, 0, stream>>>(partials, out);
}
